// Feat_GNN_70325794505146
// MI455X (gfx1250) — matrix operand order repaired and checked by reading, not compiled, not run on hardware
//
#include <hip/hip_runtime.h>
#include <hip/hip_bf16.h>

// ---------------- problem constants ----------------
#define B_      64
#define N_OBJ   32
#define N_TIME  1024
#define N_IN    6
#define ETYPES  4
#define MSG_HID 256
#define MSG_OUT 32
#define N_HID   256
#define N_OUT   8
#define OUT_CH  32
#define OUT_T   16
#define E_      992           // N_OBJ*(N_OBJ-1)
#define NROW    2048          // B_*N_OBJ

typedef __attribute__((ext_vector_type(16))) _Float16 v16h;
typedef __attribute__((ext_vector_type(8)))  float    v8f;

union HF16 { v16h v; unsigned u[8]; };

// ---------------- workspace layout (bytes) ----------------
// region A (x0, 64MB) is dead after the transpose kernel; aug/h1 overlay it.
static const size_t OFF_X0  = 0;                       // [2048][32][256] f32 = 64 MiB
static const size_t OFF_X1  = 67108864;                // [2048][32][64]  f32 = 16 MiB
static const size_t OFF_X2  = 83886080;                // [2048][32][16]  f32 = 4 MiB
static const size_t OFF_RED = 88080384;                // [64][16][32][32] f32 = 4 MiB
static const size_t OFF_AGG = 92274688;                // [64][16][32][32] f32 = 4 MiB
static const size_t OFF_W1S = 96468992;                // swizzled w1 fp16: 128 KiB
static const size_t OFF_W2S = 96600064;                // swizzled w2 fp16: 64 KiB
static const size_t OFF_AUG = 0;                       // [64][32768] f32 = 8 MiB (overlays x0)
static const size_t OFF_H1  = 8388608;                 // [64][256]  f32 (overlays x0)

// ---------------- weight swizzle into WMMA B-fragment order ----------------
// in : w [4 types][Kfull][N] f32
// out: ws[(((i*KT+kt)*NT+nt)*256) + v*32 + lane] = pack(f16(w[kb]), f16(w[kb+1]))
//      with n = nt*16 + lane%16, kb = kt*32 + (v<4 ? 2v : 16+2(v-4)) + 8*(lane/16)   (ISA B layout)
__global__ void swz_weights(const float* __restrict__ w, unsigned* __restrict__ ws,
                            int KT, int NT, int N, int Kfull)
{
    int id = blockIdx.x * blockDim.x + threadIdx.x;
    int total = ETYPES * KT * NT * 256;
    if (id >= total) return;
    int lane = id & 31;
    int v    = (id >> 5) & 7;
    int nt   = (id >> 8) % NT;
    int kt   = ((id >> 8) / NT) % KT;
    int i    = (id >> 8) / (NT * KT);
    int n  = nt * 16 + (lane & 15);
    int kb = kt * 32 + (v < 4 ? 2 * v : 16 + 2 * (v - 4)) + 8 * (lane >> 4);
    float f0 = w[(i * Kfull + kb)     * N + n];
    float f1 = w[(i * Kfull + kb + 1) * N + n];
    union { unsigned u; _Float16 h[2]; } pk;
    pk.h[0] = (_Float16)f0; pk.h[1] = (_Float16)f1;
    ws[id] = pk.u;
}

// ---------------- dilated conv block (+relu+maxpool4), f32 direct ----------------
// x strided (sn,sc,st); w [4][8][Cin][3]; y [N][32][L/4] channel-major
__global__ void dilated_conv_block(const float* __restrict__ x, const float* __restrict__ w,
                                   const float* __restrict__ bias, float* __restrict__ y,
                                   int Cin, int L, long sn, int sc, int st, int total)
{
    int idx = blockIdx.x * blockDim.x + threadIdx.x;
    if (idx >= total) return;
    const int Lout = L >> 2;
    const int tout = idx % Lout;
    const int oc   = (idx / Lout) & 31;
    const int n    = idx / (Lout * 32);
    const int br = oc >> 3, co = oc & 7;
    const int d  = 1 << br;                       // dilations 1,2,4,8
    const float* wp = w + (long)(br * 8 + co) * Cin * 3;
    const float  bb = bias[br * 8 + co];
    const float* xb = x + (long)n * sn;
    float best = -3.4e38f;
    for (int p = 0; p < 4; ++p) {
        const int t = tout * 4 + p;
        float acc = bb;
        for (int k = 0; k < 3; ++k) {
            const int tt = t + (k - 1) * d;
            if ((unsigned)tt < (unsigned)L) {
                const float* xc = xb + (long)tt * st;
                for (int c = 0; c < Cin; ++c)
                    acc += xc[c * sc] * wp[c * 3 + k];
            }
        }
        best = fmaxf(best, acc);
    }
    y[((long)n * 32 + oc) * Lout + tout] = fmaxf(best, 0.f);   // relu∘max == max∘relu
}

// ---------------- transpose to reduced[b][t][o][f] ----------------
__global__ void make_reduced(const float* __restrict__ x2, float* __restrict__ red)
{
    int idx = blockIdx.x * blockDim.x + threadIdx.x;       // 1,048,576 threads
    if (idx >= B_ * OUT_T * N_OBJ * OUT_CH) return;
    int f = idx & 31;
    int o = (idx >> 5) & 31;
    int t = (idx >> 10) & 15;
    int b = idx >> 14;
    red[idx] = x2[(((long)(b * 32 + o) * 32 + f) * 16) + t];
}

// ---------------- fused edge-MLP message kernel (WMMA f16 + async LDS DMA) ----
// One wave handles 16 edge rows of one (b,t) slice: DMAs pre_msg rows into LDS
// with GLOBAL_LOAD_ASYNC_TO_LDS_B128 (ASYNCcnt), builds A-frags, runs
// (64->256 relu)(256->32 relu) for 4 edge types with v_wmma_f32_16x16x32_f16,
// scales by rel_type, atomically aggregates per-receiver.
__global__ __launch_bounds__(64) void msg_mlp_kernel(
    const float*    __restrict__ reduced,   // [B][16][32][32]
    const unsigned* __restrict__ w1s,       // swizzled [4][2][16][256]
    const float*    __restrict__ b1,        // [4][256]
    const unsigned* __restrict__ w2s,       // swizzled [4][8][2][256]
    const float*    __restrict__ b2,        // [4][32]
    const float*    __restrict__ rel_type,  // [B][992][4]
    float*          __restrict__ agg)       // [B][16][32][32]
{
    __shared__ __align__(16) float    AfF[2][16][64];   // pre_msg rows (f32), async-DMA'd
    __shared__ __align__(16) _Float16 Hf[2][16][256];   // hidden per wave

    const int wv   = threadIdx.x >> 5;
    const int lane = threadIdx.x & 31;
    const int blk  = blockIdx.x;                         // B*16*31 blocks
    const int b    = blk / (OUT_T * 31);
    const int rt_  = blk % (OUT_T * 31);
    const int t    = rt_ / 31;
    const int c2   = rt_ % 31;
    const int e0   = (c2 * 2 + wv) * 16;

    __builtin_prefetch(w1s, 0, 0);                       // -> global_prefetch_b8

    // ---- async DMA 16 rows of pre_msg = [recv f32 x32 | send f32 x32] into LDS.
    //      16 rows x 256B = 4KB per wave = 8 iterations x 32 lanes x 16B.
    //      VDST carries the LDS byte offset (== low 32 bits of the flat pointer,
    //      the hardware adds it to the wave's LDS base).
    {
        #pragma unroll
        for (int it = 0; it < 8; ++it) {
            const int linear = it * 32 + lane;           // 0..255
            const int row = linear >> 4;                 // 0..15
            const int ck  = linear & 15;                 // 16B chunk within 256B row
            const int e = e0 + row;
            const int r = e / 31;
            const int j = e - r * 31;
            const int s = j + (j >= r ? 1 : 0);
            const int obj = (ck < 8) ? r : s;
            const float* gsrc = reduced + (((b * OUT_T + t) * N_OBJ + obj) << 5)
                                        + ((ck & 7) << 2);
            unsigned ldsoff = (unsigned)(unsigned long long)(const void*)
                                  &AfF[wv][row][ck << 2];
            asm volatile("global_load_async_to_lds_b128 %0, %1, off"
                         :: "v"(ldsoff), "v"((unsigned long long)gsrc)
                         : "memory");
        }
        asm volatile("s_wait_asynccnt 0" ::: "memory");
    }
    __syncthreads();

    const int m    = lane & 15;      // A row / D column within tile
    const int hh   = lane >> 4;

    // ---- build the two K-tile A fragments (ISA 16-bit 16x32 A layout) ----
    HF16 a0, a1;
    #pragma unroll
    for (int v = 0; v < 8; ++v) {
        const int kk = (v < 4 ? 2 * v : 16 + 2 * (v - 4)) + 8 * hh;
        union { unsigned u; _Float16 h[2]; } p0, p1;
        p0.h[0] = (_Float16)AfF[wv][m][kk];
        p0.h[1] = (_Float16)AfF[wv][m][kk + 1];
        p1.h[0] = (_Float16)AfF[wv][m][kk + 32];
        p1.h[1] = (_Float16)AfF[wv][m][kk + 33];
        a0.u[v] = p0.u;
        a1.u[v] = p1.u;
    }

    float outAcc[2][8];
    #pragma unroll
    for (int q = 0; q < 2; ++q)
        #pragma unroll
        for (int v = 0; v < 8; ++v) outAcc[q][v] = 0.f;

    for (int ty = 0; ty < ETYPES; ++ty) {
        __syncthreads();     // previous type's layer2 reads done before Hf rewrite
        // ---- layer 1: [16x64] x [64x256] -> relu -> Hf ----
        #pragma unroll 4
        for (int nt = 0; nt < 16; ++nt) {
            v8f acc = {};
            #pragma unroll
            for (int kt = 0; kt < 2; ++kt) {
                HF16 bf;
                const unsigned* wp = w1s + (((ty * 2 + kt) * 16 + nt) << 8) + lane;
                #pragma unroll
                for (int v = 0; v < 8; ++v) bf.u[v] = wp[v << 5];
                acc = __builtin_amdgcn_wmma_f32_16x16x32_f16(
                    false, (kt == 0 ? a0.v : a1.v), false, bf.v,
                    (short)0, acc, false, false);
            }
            const float bias = b1[ty * MSG_HID + nt * 16 + m];
            #pragma unroll
            for (int v = 0; v < 8; ++v) {
                float h = acc[v] + bias;
                h = h > 0.f ? h : 0.f;
                Hf[wv][v + 8 * hh][nt * 16 + m] = (_Float16)h;
            }
        }
        __syncthreads();
        // ---- layer 2: [16x256] x [256x32] -> relu -> *rel_type, accumulate ----
        #pragma unroll
        for (int nt = 0; nt < 2; ++nt) {
            v8f acc = {};
            for (int kt = 0; kt < 8; ++kt) {
                HF16 af, bf;
                #pragma unroll
                for (int v = 0; v < 8; ++v) {
                    const int kk = kt * 32 + (v < 4 ? 2 * v : 16 + 2 * (v - 4)) + 8 * hh;
                    af.u[v] = *(const unsigned*)&Hf[wv][m][kk];
                }
                const unsigned* wp = w2s + (((ty * 8 + kt) * 2 + nt) << 8) + lane;
                #pragma unroll
                for (int v = 0; v < 8; ++v) bf.u[v] = wp[v << 5];
                acc = __builtin_amdgcn_wmma_f32_16x16x32_f16(
                    false, af.v, false, bf.v, (short)0, acc, false, false);
            }
            const float bias2 = b2[ty * MSG_OUT + nt * 16 + m];
            #pragma unroll
            for (int v = 0; v < 8; ++v) {
                float h = acc[v] + bias2;
                h = h > 0.f ? h : 0.f;
                const int e = e0 + v + 8 * hh;
                const float rt = rel_type[((long)b * E_ + e) * ETYPES + ty];
                outAcc[nt][v] += h * rt;
            }
        }
    }

    // ---- aggregate per receiver: agg[b][t][r][:] += msgs ----
    #pragma unroll
    for (int nt = 0; nt < 2; ++nt)
        #pragma unroll
        for (int v = 0; v < 8; ++v) {
            const int e = e0 + v + 8 * hh;
            const int r = e / 31;
            atomicAdd(&agg[(((b * OUT_T + t) * N_OBJ + r) << 5) + nt * 16 + m],
                      outAcc[nt][v]);
        }
}

// ---------------- pack aug[b][ (t*32+o)*64 + f ] ----------------
__global__ void pack_aug(const float* __restrict__ red, const float* __restrict__ agg,
                         float* __restrict__ aug)
{
    int idx = blockIdx.x * blockDim.x + threadIdx.x;   // 64*32768
    if (idx >= B_ * 32768) return;
    int b = idx >> 15;
    int r = idx & 32767;
    int t = r >> 11;
    int o = (r >> 6) & 31;
    int f = r & 63;
    const float* src = (f < 32) ? red : agg;
    aug[idx] = src[(((b * OUT_T + t) * N_OBJ + o) << 5) + (f & 31)];
}

// ---------------- FC head (f32 VALU; ~1 GFLOP, negligible) ----------------
__global__ __launch_bounds__(256) void fc1_kernel(const float* __restrict__ aug,
                                                  const float* __restrict__ w,
                                                  const float* __restrict__ bias,
                                                  float* __restrict__ out1)
{
    __shared__ float red[256];
    const int bb = blockIdx.x >> 8;
    const int j  = blockIdx.x & 255;
    float s = 0.f;
    for (int k = threadIdx.x; k < 32768; k += 256)
        s += aug[(long)bb * 32768 + k] * w[(long)k * 256 + j];
    red[threadIdx.x] = s;
    __syncthreads();
    for (int off = 128; off > 0; off >>= 1) {
        if (threadIdx.x < off) red[threadIdx.x] += red[threadIdx.x + off];
        __syncthreads();
    }
    if (threadIdx.x == 0) {
        float v = red[0] + bias[j];
        out1[bb * 256 + j] = v > 0.f ? v : 0.f;
    }
}

__global__ void fc2_kernel(const float* __restrict__ h1, const float* __restrict__ w,
                           const float* __restrict__ bias, float* __restrict__ out)
{
    int tid = blockIdx.x * blockDim.x + threadIdx.x;   // 512 = 64*8
    if (tid >= B_ * N_OUT) return;
    int b = tid >> 3, j = tid & 7;
    float s = bias[j];
    for (int k = 0; k < 256; ++k)
        s += h1[b * 256 + k] * w[k * 8 + j];
    out[tid] = s;
}

// ---------------- launcher ----------------
extern "C" void kernel_launch(void* const* d_in, const int* in_sizes, int n_in,
                              void* d_out, int out_size, void* d_ws, size_t ws_size,
                              hipStream_t stream)
{
    const float* inputs   = (const float*)d_in[0];
    const float* rel_type = (const float*)d_in[1];
    // d_in[2]=rel_rec, d_in[3]=rel_send: one-hot, replaced by analytic indices
    const float* cnn_w0 = (const float*)d_in[4];
    const float* cnn_b0 = (const float*)d_in[5];
    const float* cnn_w1 = (const float*)d_in[6];
    const float* cnn_b1 = (const float*)d_in[7];
    const float* cnn_w2 = (const float*)d_in[8];
    const float* cnn_b2 = (const float*)d_in[9];
    const float* msg_w1 = (const float*)d_in[10];
    const float* msg_b1 = (const float*)d_in[11];
    const float* msg_w2 = (const float*)d_in[12];
    const float* msg_b2 = (const float*)d_in[13];
    const float* fc1_w  = (const float*)d_in[14];
    const float* fc1_b  = (const float*)d_in[15];
    const float* fc2_w  = (const float*)d_in[16];
    const float* fc2_b  = (const float*)d_in[17];

    char* ws = (char*)d_ws;
    float*    x0   = (float*)(ws + OFF_X0);
    float*    x1   = (float*)(ws + OFF_X1);
    float*    x2   = (float*)(ws + OFF_X2);
    float*    red  = (float*)(ws + OFF_RED);
    float*    agg  = (float*)(ws + OFF_AGG);
    unsigned* w1s  = (unsigned*)(ws + OFF_W1S);
    unsigned* w2s  = (unsigned*)(ws + OFF_W2S);
    float*    aug  = (float*)(ws + OFF_AUG);   // overlays x0 (dead by then)
    float*    h1   = (float*)(ws + OFF_H1);    // overlays x0

    // weight swizzle + agg zero
    swz_weights<<<(ETYPES * 2 * 16 * 256) / 256, 256, 0, stream>>>(msg_w1, w1s, 2, 16, MSG_HID, 64);
    swz_weights<<<(ETYPES * 8 * 2 * 256) / 256, 256, 0, stream>>>(msg_w2, w2s, 8, 2, MSG_OUT, 256);
    hipMemsetAsync(agg, 0, (size_t)B_ * OUT_T * N_OBJ * OUT_CH * sizeof(float), stream);

    // CNN encoder: [2048,6,1024] -> [2048,32,256] -> [2048,32,64] -> [2048,32,16]
    {
        int total0 = NROW * 32 * 256;
        dilated_conv_block<<<total0 / 256, 256, 0, stream>>>(
            inputs, cnn_w0, cnn_b0, x0, N_IN, 1024, (long)1024 * N_IN, 1, N_IN, total0);
        int total1 = NROW * 32 * 64;
        dilated_conv_block<<<total1 / 256, 256, 0, stream>>>(
            x0, cnn_w1, cnn_b1, x1, 32, 256, (long)32 * 256, 256, 1, total1);
        int total2 = NROW * 32 * 16;
        dilated_conv_block<<<total2 / 256, 256, 0, stream>>>(
            x1, cnn_w2, cnn_b2, x2, 32, 64, (long)32 * 64, 64, 1, total2);
    }

    // transpose into reduced[b][t][o][f]
    make_reduced<<<(B_ * OUT_T * N_OBJ * OUT_CH) / 256, 256, 0, stream>>>(x2, red);

    // fused WMMA message passing: B*OUT_T*31 blocks, 2 waves each (62 chunks of 16 edges)
    msg_mlp_kernel<<<B_ * OUT_T * 31, 64, 0, stream>>>(
        red, w1s, msg_b1, w2s, msg_b2, rel_type, agg);

    // FC head
    pack_aug<<<(B_ * 32768) / 256, 256, 0, stream>>>(red, agg, aug);
    fc1_kernel<<<B_ * N_HID, 256, 0, stream>>>(aug, fc1_w, fc1_b, h1);
    fc2_kernel<<<2, 256, 0, stream>>>(h1, fc2_w, fc2_b, (float*)d_out);
}
